// Pointnet2MSG_Cls_44409961840940
// MI455X (gfx1250) — compile-verified, matrix operand order checked by reading, not run on hardware
//
#include <hip/hip_runtime.h>
#include <cstdint>
#include <cstddef>

// ---------------------------------------------------------------------------
// PointNet++ MSG classifier forward, MI455X (gfx1250, wave32, WMMA).
// Pipeline: transpose -> FPS -> ball query -> [gather -> WMMA GEMM(+BN+ReLU)
// x3 -> seg-max] per scale -> global SA -> FC head.
// Matrix math: v_wmma_f32_16x16x32_f16; weights pre-transposed to f16
// [Cout][Cin_pad32]; BN folded to y*a+c. GEMM N-subtiling is a template
// constant so accumulators stay in VGPRs (no spills, no exec juggling).
// ---------------------------------------------------------------------------

typedef _Float16 half_t;
typedef _Float16 v16h __attribute__((ext_vector_type(16)));
typedef float    v8f  __attribute__((ext_vector_type(8)));

static inline int cdiv_h(int a, int b) { return (a + b - 1) / b; }
static inline size_t alignup_h(size_t x, size_t a) { return (x + a - 1) & ~(a - 1); }

#define BN_EPS 1e-5f

// =========================== device kernels ================================

// [B,3,N] -> [B,N,3]
__global__ void k_transpose(const float* __restrict__ pts, float* __restrict__ xyz, int n) {
  int t = blockIdx.x * blockDim.x + threadIdx.x;
  if (t >= 8 * n) return;
  int b = t / n, j = t % n;
  const float* p = pts + (size_t)b * 3 * n;
  float* o = xyz + (size_t)t * 3;
  o[0] = p[j];
  o[1] = p[n + j];
  o[2] = p[2 * n + j];
}

// Iterative farthest point sampling, one workgroup per batch.
// dist[] lives in LDS (n <= 8192 -> 32KB). First pick is index 0 (ref semantics).
__global__ void k_fps(const float* __restrict__ xyz, int n, int np,
                      int* __restrict__ fidx, float* __restrict__ nxyz) {
  __shared__ float s_dist[8192];
  __shared__ float s_rv[512];
  __shared__ int   s_ri[512];
  int b = blockIdx.x, t = threadIdx.x, bs = blockDim.x;
  const float* base = xyz + (size_t)b * n * 3;
  for (int j = t; j < n; j += bs) s_dist[j] = 1e10f;
  if (t == 0) {
    fidx[b * np] = 0;
    nxyz[(size_t)b * np * 3 + 0] = base[0];
    nxyz[(size_t)b * np * 3 + 1] = base[1];
    nxyz[(size_t)b * np * 3 + 2] = base[2];
  }
  int last = 0;
  __syncthreads();
  for (int i = 1; i < np; ++i) {
    float lx = base[last * 3], ly = base[last * 3 + 1], lz = base[last * 3 + 2];
    float bd = -1.0f; int bi = 0;
    for (int j = t; j < n; j += bs) {
      float dx = base[j * 3] - lx, dy = base[j * 3 + 1] - ly, dz = base[j * 3 + 2] - lz;
      float d = fminf(s_dist[j], dx * dx + dy * dy + dz * dz);
      s_dist[j] = d;
      if (d > bd) { bd = d; bi = j; }   // strided j ascending -> first-max per thread
    }
    s_rv[t] = bd; s_ri[t] = bi;
    __syncthreads();
    for (int off = bs >> 1; off > 0; off >>= 1) {
      if (t < off) {
        float ov = s_rv[t + off]; int oi = s_ri[t + off];
        if (ov > s_rv[t] || (ov == s_rv[t] && oi < s_ri[t])) { s_rv[t] = ov; s_ri[t] = oi; }
      }
      __syncthreads();
    }
    last = s_ri[0];
    if (t == 0) {
      fidx[b * np + i] = last;
      nxyz[((size_t)b * np + i) * 3 + 0] = base[last * 3];
      nxyz[((size_t)b * np + i) * 3 + 1] = base[last * 3 + 1];
      nxyz[((size_t)b * np + i) * 3 + 2] = base[last * 3 + 2];
    }
    __syncthreads();
  }
}

// First K in-range indices in ascending order, pad with first hit.
__global__ void k_ball_query(const float* __restrict__ xyz, const float* __restrict__ ctr,
                             int* __restrict__ out, int n, int S, int K, float r2, int total) {
  int cg = blockIdx.x * blockDim.x + threadIdx.x;
  if (cg >= total) return;
  int b = cg / S;
  const float* pc = ctr + (size_t)cg * 3;
  float cx = pc[0], cy = pc[1], cz = pc[2];
  const float* base = xyz + (size_t)b * n * 3;
  int* o = out + (size_t)cg * K;
  int cnt = 0;
  for (int j = 0; j < n && cnt < K; ++j) {
    float dx = base[j * 3] - cx, dy = base[j * 3 + 1] - cy, dz = base[j * 3 + 2] - cz;
    if (dx * dx + dy * dy + dz * dz < r2) o[cnt++] = j;
  }
  int f = (cnt > 0) ? o[0] : 0;
  for (; cnt < K; ++cnt) o[cnt] = f;
}

// Build gathered f16 input rows [rows, lda]: rel-xyz (if ctr) + feats + zero pad.
// idx==null -> p=j (group_all); ctr==null -> no recentering.
__global__ void k_build_x0(const float* __restrict__ xyz, const float* __restrict__ ctr,
                           const float* __restrict__ feats, const int* __restrict__ idx,
                           half_t* __restrict__ X0, int n, int S, int K, int Cf, int lda,
                           int c0, int nrows) {
  int row = blockIdx.x * blockDim.x + threadIdx.x;
  if (row >= nrows) return;
  int ci = row / K, j = row % K;
  int cg = c0 + ci;
  int b = cg / S, s = cg % S;
  int p = idx ? idx[(size_t)cg * K + j] : j;
  const float* pp = xyz + ((size_t)b * n + p) * 3;
  float cx = 0.f, cy = 0.f, cz = 0.f;
  if (ctr) { const float* pc = ctr + ((size_t)b * S + s) * 3; cx = pc[0]; cy = pc[1]; cz = pc[2]; }
  half_t* o = X0 + (size_t)row * lda;
  o[0] = (half_t)(pp[0] - cx);
  o[1] = (half_t)(pp[1] - cy);
  o[2] = (half_t)(pp[2] - cz);
  int ch = 3;
  if (feats) {
    const float* pf = feats + ((size_t)b * n + p) * Cf;
    for (int f = 0; f < Cf; ++f) o[ch++] = (half_t)pf[f];
  }
  for (; ch < lda; ++ch) o[ch] = (half_t)0.f;
}

// W f32 [cin,cout] -> Wt f16 [cout, cin_pad] (transposed + zero padded);
// also folds BN: a = gamma*rsqrt(var+eps), c = beta - mean*a.
__global__ void k_prep_w(const float* __restrict__ W, const float* __restrict__ g,
                         const float* __restrict__ be, const float* __restrict__ mu,
                         const float* __restrict__ var, half_t* __restrict__ Wt,
                         float* __restrict__ a, float* __restrict__ c,
                         int cin, int cout, int cinp) {
  int t = blockIdx.x * blockDim.x + threadIdx.x;
  int co = t / cinp, ci = t % cinp;
  if (co >= cout) return;
  Wt[(size_t)co * cinp + ci] = (ci < cin) ? (half_t)W[(size_t)ci * cout + co] : (half_t)0.f;
  if (ci == 0) {
    float ai = g[co] * rsqrtf(var[co] + BN_EPS);
    a[co] = ai;
    c[co] = be[co] - mu[co] * ai;
  }
}

// Per-lane WMMA fragment loader. Both A [rows,lda] and Wt [cout,lda] are
// row-major with the "lane = row, 8+8 K-halves per lane" ISA layout:
//  lane<16 : row = row0+lane,    halves k0..k0+7 then k0+16..k0+23
//  lane>=16: row = row0+lane-16, halves k0+8..k0+15 then k0+24..k0+31
union FragU { v16h v; uint4 q[2]; };
__device__ __forceinline__ v16h load_frag(const half_t* __restrict__ base,
                                          int row0, int lda, int k0, int lane) {
  const half_t* p = base + (size_t)(row0 + (lane & 15)) * lda + (k0 + ((lane >> 4) << 3));
  FragU f;
  f.q[0] = *(const uint4*)p;
  f.q[1] = *(const uint4*)(p + 16);
  return f.v;
}

// D = relu(bn(A @ Wt^T)) in f16. rows % 16 == 0, lda % 32 == 0,
// (cout/16) % NSUB == 0. Each wave owns a 16 x (16*NSUB) output tile;
// NSUB is compile-time so accumulators never leave VGPRs and there is no
// control flow around the WMMAs (EXEC stays all-ones).
template <int NSUB>
__global__ void __launch_bounds__(256)
k_gemm_bn_relu(const half_t* __restrict__ A, const half_t* __restrict__ Wt,
               const float* __restrict__ aa, const float* __restrict__ cc,
               half_t* __restrict__ out, int rows, int lda, int cout) {
  int wave = (int)((blockIdx.x * blockDim.x + threadIdx.x) >> 5);
  int lane = threadIdx.x & 31;
  int ng = (cout >> 4) / NSUB;            // N-tile groups
  int tiles = (rows >> 4) * ng;
  if (wave >= tiles) return;              // wave-uniform
  int mi = wave / ng, gi = wave % ng;
  int m0 = mi << 4;
  int n0b = gi * (NSUB << 4);

  v8f acc[NSUB] = {};
  for (int k0 = 0; k0 < lda; k0 += 32) {
    v16h af = load_frag(A, m0, lda, k0, lane);
#pragma unroll
    for (int s = 0; s < NSUB; ++s) {
      v16h bf = load_frag(Wt, n0b + (s << 4), lda, k0, lane);
      acc[s] = __builtin_amdgcn_wmma_f32_16x16x32_f16(
          false, af, false, bf, (short)0, acc[s], false, false);
    }
  }
  // D layout: lanes 0-15 -> N=lane, M=v; lanes 16-31 -> N=lane-16, M=v+8.
  int Mb = (lane >> 4) << 3;
  int N = lane & 15;
#pragma unroll
  for (int s = 0; s < NSUB; ++s) {
    int n0 = n0b + (s << 4);
    float sa = aa[n0 + N], sc = cc[n0 + N];
#pragma unroll
    for (int v = 0; v < 8; ++v) {
      float y = acc[s][v] * sa + sc;
      y = y > 0.f ? y : 0.f;
      out[(size_t)(m0 + Mb + v) * cout + n0 + N] = (half_t)y;
    }
  }
}

// Column-wise max over each group of K rows -> feats[(c0+ci)*Ctot + off + ch] (f32).
__global__ void k_seg_max(const half_t* __restrict__ Y, float* __restrict__ feats,
                          int K, int C, int c0, int nc, int Ctot, int off) {
  int t = blockIdx.x * blockDim.x + threadIdx.x;
  if (t >= nc * C) return;
  int ci = t / C, ch = t % C;
  const half_t* p = Y + (size_t)ci * K * C + ch;
  float m = (float)p[0];
  for (int j = 1; j < K; ++j) m = fmaxf(m, (float)p[(size_t)j * C]);
  feats[(size_t)(c0 + ci) * Ctot + off + ch] = m;
}

// Tiny FC head (rows = 8): mode 0 = BN+ReLU, mode 1 = +bias only.
__global__ void k_fc(const float* __restrict__ x, const float* __restrict__ W,
                     const float* __restrict__ g, const float* __restrict__ be,
                     const float* __restrict__ mu, const float* __restrict__ var,
                     const float* __restrict__ bias, float* __restrict__ out,
                     int rows, int cin, int cout, int mode) {
  int t = blockIdx.x * blockDim.x + threadIdx.x;
  if (t >= rows * cout) return;
  int r = t / cout, co = t % cout;
  float acc = 0.f;
  const float* xr = x + (size_t)r * cin;
  for (int k = 0; k < cin; ++k) acc += xr[k] * W[(size_t)k * cout + co];
  if (mode == 0) {
    float a = g[co] * rsqrtf(var[co] + BN_EPS);
    out[t] = fmaxf(0.f, (acc - mu[co]) * a + be[co]);
  } else {
    out[t] = acc + bias[co];
  }
}

// =============================== host =====================================

extern "C" void kernel_launch(void* const* d_in, const int* in_sizes, int n_in,
                              void* d_out, int out_size, void* d_ws, size_t ws_size,
                              hipStream_t stream) {
  (void)in_sizes; (void)n_in; (void)out_size; (void)ws_size;
  const int B = 8, N = 8192, S1 = 512, S2 = 128, CH = 256;

  const float* pts = (const float*)d_in[0];

  // ---- walk the parameter pytree (insertion order: W,gamma,beta,mean,var) ----
  struct MLPLayer {
    const float *W, *g, *be, *mu, *var;
    int cin, cout, cinp;
    half_t* wt; float *a, *c;
  };
  MLPLayer L[21];
  int ii = 1, nl = 0;
  auto take = [&](int cin, int cout) {
    MLPLayer& l = L[nl++];
    l.W  = (const float*)d_in[ii++];
    l.g  = (const float*)d_in[ii++];
    l.be = (const float*)d_in[ii++];
    l.mu = (const float*)d_in[ii++];
    l.var = (const float*)d_in[ii++];
    l.cin = cin; l.cout = cout; l.cinp = (cin + 31) & ~31;
  };
  const int D1[3][4] = {{3,32,32,64},{3,64,64,128},{3,64,96,128}};
  const int D2[3][4] = {{323,64,64,128},{323,128,128,256},{323,128,128,256}};
  const int DG[4]    = {643,256,512,1024};
  for (int s = 0; s < 3; ++s) for (int l = 0; l < 3; ++l) take(D1[s][l], D1[s][l + 1]);
  for (int s = 0; s < 3; ++s) for (int l = 0; l < 3; ++l) take(D2[s][l], D2[s][l + 1]);
  for (int l = 0; l < 3; ++l) take(DG[l], DG[l + 1]);
  // fc[0]: 1024->512, fc[1]: 512->256, final 256->40
  const float* fW0 = (const float*)d_in[ii++]; const float* fg0 = (const float*)d_in[ii++];
  const float* fb0 = (const float*)d_in[ii++]; const float* fm0 = (const float*)d_in[ii++];
  const float* fv0 = (const float*)d_in[ii++];
  const float* fW1 = (const float*)d_in[ii++]; const float* fg1 = (const float*)d_in[ii++];
  const float* fb1 = (const float*)d_in[ii++]; const float* fm1 = (const float*)d_in[ii++];
  const float* fv1 = (const float*)d_in[ii++];
  const float* fWf = (const float*)d_in[ii++]; const float* fbf = (const float*)d_in[ii++];

  // ---- workspace bump allocator (~55 MB total) ----
  char* w = (char*)d_ws;
  auto alloc = [&](size_t bytes) -> void* {
    void* p = (void*)w;
    w += alignup_h(bytes, 256);
    return p;
  };
  float* xyz1   = (float*)alloc((size_t)B * N * 3 * 4);
  int*   fidx1  = (int*)alloc((size_t)B * S1 * 4);
  float* nxyz1  = (float*)alloc((size_t)B * S1 * 3 * 4);
  int*   fidx2  = (int*)alloc((size_t)B * S2 * 4);
  float* nxyz2  = (float*)alloc((size_t)B * S2 * 3 * 4);
  float* feats1 = (float*)alloc((size_t)B * S1 * 320 * 4);
  float* feats2 = (float*)alloc((size_t)B * S2 * 640 * 4);
  float* gfeat  = (float*)alloc((size_t)B * 1024 * 4);
  float* xb1    = (float*)alloc((size_t)B * 512 * 4);
  float* xb2    = (float*)alloc((size_t)B * 256 * 4);
  int*   idxbuf = (int*)alloc((size_t)B * S1 * 128 * 4);
  for (int i = 0; i < 21; ++i) {
    L[i].wt = (half_t*)alloc((size_t)L[i].cout * L[i].cinp * 2);
    L[i].a  = (float*)alloc((size_t)L[i].cout * 4);
    L[i].c  = (float*)alloc((size_t)L[i].cout * 4);
  }
  half_t* bufA = (half_t*)alloc((size_t)(CH * 128) * 352 * 2);  // X0 / even activations
  half_t* bufB = (half_t*)alloc((size_t)(CH * 128) * 256 * 2);  // odd activations (global: 1024 rows x 1024 cols)

  // ---- weight prep (f32 -> transposed padded f16, folded BN) ----
  for (int i = 0; i < 21; ++i) {
    int tot = L[i].cout * L[i].cinp;
    k_prep_w<<<cdiv_h(tot, 256), 256, 0, stream>>>(
        L[i].W, L[i].g, L[i].be, L[i].mu, L[i].var,
        L[i].wt, L[i].a, L[i].c, L[i].cin, L[i].cout, L[i].cinp);
  }

  auto gemm = [&](const half_t* A, const MLPLayer& Ld, half_t* out, int rows) {
    int nt = Ld.cout >> 4;
    if ((nt & 3) == 0) {
      int tiles = (rows >> 4) * (nt >> 2);
      k_gemm_bn_relu<4><<<cdiv_h(tiles, 8), 256, 0, stream>>>(
          A, Ld.wt, Ld.a, Ld.c, out, rows, Ld.cinp, Ld.cout);
    } else {                                   // cout = 32 or 96
      int tiles = (rows >> 4) * (nt >> 1);
      k_gemm_bn_relu<2><<<cdiv_h(tiles, 8), 256, 0, stream>>>(
          A, Ld.wt, Ld.a, Ld.c, out, rows, Ld.cinp, Ld.cout);
    }
  };

  // ---- stage 0: layout + sampling ----
  k_transpose<<<cdiv_h(B * N, 256), 256, 0, stream>>>(pts, xyz1, N);
  k_fps<<<B, 512, 0, stream>>>(xyz1, N, S1, fidx1, nxyz1);

  // ---- SA layer 1 (S=512, feats=None) ----
  {
    const float R1[3] = {0.1f, 0.2f, 0.4f};
    const int   K1[3] = {16, 32, 128};
    int off = 0;
    for (int sc = 0; sc < 3; ++sc) {
      int K = K1[sc]; float r2 = R1[sc] * R1[sc];
      k_ball_query<<<cdiv_h(B * S1, 128), 128, 0, stream>>>(
          xyz1, nxyz1, idxbuf, N, S1, K, r2, B * S1);
      MLPLayer* m = &L[sc * 3];
      for (int c0 = 0; c0 < B * S1; c0 += CH) {
        int rows = CH * K;
        k_build_x0<<<cdiv_h(rows, 128), 128, 0, stream>>>(
            xyz1, nxyz1, nullptr, idxbuf, bufA, N, S1, K, 0, m[0].cinp, c0, rows);
        gemm(bufA, m[0], bufB, rows);
        gemm(bufB, m[1], bufA, rows);
        gemm(bufA, m[2], bufB, rows);
        k_seg_max<<<cdiv_h(CH * m[2].cout, 256), 256, 0, stream>>>(
            bufB, feats1, K, m[2].cout, c0, CH, 320, off);
      }
      off += m[2].cout;
    }
  }

  // ---- SA layer 2 (S=128, over the 512 layer-1 centers) ----
  k_fps<<<B, 512, 0, stream>>>(nxyz1, S1, S2, fidx2, nxyz2);
  {
    const float R2[3] = {0.2f, 0.4f, 0.8f};
    const int   K2[3] = {32, 64, 128};
    int off = 0;
    for (int sc = 0; sc < 3; ++sc) {
      int K = K2[sc]; float r2 = R2[sc] * R2[sc];
      k_ball_query<<<cdiv_h(B * S2, 128), 128, 0, stream>>>(
          nxyz1, nxyz2, idxbuf, S1, S2, K, r2, B * S2);
      MLPLayer* m = &L[9 + sc * 3];
      for (int c0 = 0; c0 < B * S2; c0 += CH) {
        int rows = CH * K;
        k_build_x0<<<cdiv_h(rows, 128), 128, 0, stream>>>(
            nxyz1, nxyz2, feats1, idxbuf, bufA, S1, S2, K, 320, m[0].cinp, c0, rows);
        gemm(bufA, m[0], bufB, rows);
        gemm(bufB, m[1], bufA, rows);
        gemm(bufA, m[2], bufB, rows);
        k_seg_max<<<cdiv_h(CH * m[2].cout, 256), 256, 0, stream>>>(
            bufB, feats2, K, m[2].cout, c0, CH, 640, off);
      }
      off += m[2].cout;
    }
  }

  // ---- global SA (group_all, no recentering): rows = B*128, 643->...->1024 ----
  {
    MLPLayer* m = &L[18];
    int rows = B * 128;
    k_build_x0<<<cdiv_h(rows, 128), 128, 0, stream>>>(
        nxyz2, nullptr, feats2, nullptr, bufA, 128, 1, 128, 640, m[0].cinp, 0, rows);
    gemm(bufA, m[0], bufB, rows);
    gemm(bufB, m[1], bufA, rows);
    gemm(bufA, m[2], bufB, rows);
    k_seg_max<<<cdiv_h(B * 1024, 256), 256, 0, stream>>>(
        bufB, gfeat, 128, 1024, 0, B, 1024, 0);
  }

  // ---- FC head ----
  k_fc<<<cdiv_h(B * 512, 256), 256, 0, stream>>>(
      gfeat, fW0, fg0, fb0, fm0, fv0, nullptr, xb1, B, 1024, 512, 0);
  k_fc<<<cdiv_h(B * 256, 256), 256, 0, stream>>>(
      xb1, fW1, fg1, fb1, fm1, fv1, nullptr, xb2, B, 512, 256, 0);
  k_fc<<<cdiv_h(B * 40, 256), 256, 0, stream>>>(
      xb2, fWf, nullptr, nullptr, nullptr, nullptr, fbf, (float*)d_out, B, 256, 40, 1);
}
